// TimewarpModel_31001073943152
// MI455X (gfx1250) — matrix operand order repaired and checked by reading, not compiled, not run on hardware
//
#include <hip/hip_runtime.h>
#include <hip/hip_bf16.h>

typedef __attribute__((ext_vector_type(16))) __bf16 v16bf;
typedef __attribute__((ext_vector_type(8)))  __bf16 v8bf;
typedef __attribute__((ext_vector_type(8)))  float  v8f;
typedef unsigned int u32x4 __attribute__((ext_vector_type(4)));
typedef int          i32x8 __attribute__((ext_vector_type(8)));
typedef int          i32x4 __attribute__((ext_vector_type(4)));

#define TOK 8192
#define NBATCH 4
#define NTOK 2048
#define HID 320
#define CIW 352   // CI padded width (K=323 padded to multiple of 32)

#if defined(__gfx1250__) && __has_builtin(__builtin_amdgcn_tensor_load_to_lds)
#define HAVE_TDM 1
#else
#define HAVE_TDM 0
#endif

// =====================================================================
// bf16-WMMA GEMM: C = act(A @ B + bias), f32 accumulate.
//   A: bf16 (ABF16=1): double-buffered LDS tiles, staged by the Tensor
//      Data Mover (pipelined: tile i+1 streams while tile i computes),
//      or f32 (converted in staging).
//   B: f32 row-major KxN, converted + staged transposed (Bs[n][k]) in LDS
//   C: f32 (CBF16=0) or bf16 (CBF16=1), row-major, ldc stride
// Block = MWAVES*NWAVES wave32s; wave tile = (MFRAG x NFRAG) 16x16 tiles.
// blockIdx.z batches with element strides sA/sB/sC.
// TDM requires: M % BM == 0, A rows readable up to K rounded up to 32.
// =====================================================================
template<int MWAVES, int NWAVES, int MFRAG, int NFRAG, bool ABF16, int ACT, int CBF16>
__global__ __launch_bounds__(MWAVES*NWAVES*32)
void gemm_wmma(const void* __restrict__ Av, const float* __restrict__ Bg,
               const float* __restrict__ bias, void* __restrict__ Cg,
               int M, int N, int K, int lda, int ldb, int ldc,
               long sA, long sB, long sC)
{
  constexpr int BM = MWAVES*MFRAG*16;
  constexpr int BN = NWAVES*NFRAG*16;
  constexpr int BK = 32;
  constexpr int NT = MWAVES*NWAVES*32;
  __shared__ __bf16 As[2][BM][BK+8];  // double-buffered; row stride 40 elems
  __shared__ __bf16 Bs[BN][BK+8];     // transposed: Bs[n][k]

  const int tid  = threadIdx.x;
  const int wave = tid >> 5;
  const int lane = tid & 31;
  const int wm = wave % MWAVES;
  const int wn = wave / MWAVES;
  const int bm0 = blockIdx.y * BM;
  const int bn0 = blockIdx.x * BN;
  const int z   = blockIdx.z;

  const float*  Af  = (const float*)Av  + (ABF16 ? 0 : (long)z * sA);
  const __bf16* Ab  = (const __bf16*)Av + (ABF16 ? (long)z * sA : 0);
  const float*  Bz  = Bg + (long)z * sB;
  float*        Czf = (float*)Cg  + (CBF16 ? 0 : (long)z * sC);
  __bf16*       Czh = (__bf16*)Cg + (CBF16 ? (long)z * sC : 0);

  // stage A tile `it` into buffer `buf`
  auto stage_a = [&](int it, int buf) {
    const int k0 = it * BK;
    if (ABF16) {
#if HAVE_TDM
      // One 2D D# per tile, issued by wave 0 (TDM is a per-wave DMA engine).
      // ISA 8.3-8.5: data_size=2B; tile 32 x BM; row stride lda; LDS pad
      // 4 DWORDs after every 16 DWORDs -> 40-element LDS row stride.
      if (wave == 0) {
        unsigned long long ga =
            (unsigned long long)(const void*)(Ab + (long)bm0 * lda + k0);
        unsigned lds_off =
            (unsigned)(unsigned long long)(const void*)&As[buf][0][0];
        u32x4 g0 = { 1u /*count=1*/, lds_off,
                     (unsigned)(ga & 0xffffffffull),
                     (unsigned)((ga >> 32) & 0x1ffffffull) | (2u << 30) /*type=2*/ };
        i32x8 g1;
        g1[0] = (int)((1u << 16)    // data_size = 2 bytes
                    | (1u << 20)    // pad_enable
                    | (3u << 22)    // pad_interval: 16 DWORDs
                    | (3u << 25));  // pad_amount: 4 DWORDs
        g1[1] = 0;                          // tensor dims set huge (no OOB clip)
        g1[2] = 0x4000;
        g1[3] = 0x4000 | (32 << 16);        // tile_dim0 = 32
        g1[4] = BM;                         // tile_dim1 = BM, tile_dim2 = 0
        g1[5] = lda;                        // tensor_dim0_stride (elements)
        g1[6] = 0;
        g1[7] = 0;
        i32x4 gz4 = { 0, 0, 0, 0 };
        i32x8 gz8 = { 0, 0, 0, 0, 0, 0, 0, 0 };
        __builtin_amdgcn_tensor_load_to_lds(g0, g1, gz4, gz4, gz8, 0);
      }
#else
      for (int e = tid * 8; e < BM * BK; e += NT * 8) {
        int r = e >> 5, c = e & 31;
        int gm = bm0 + r; long gk = k0 + c;
        if (gm < M && gk + 7 < K) {
          *(uint4*)&As[buf][r][c] = *(const uint4*)(Ab + (long)gm * lda + gk);
        } else {
#pragma unroll
          for (int q = 0; q < 8; ++q) {
            __bf16 vq = (__bf16)0.f;
            if (gm < M && gk + q < K) vq = Ab[(long)gm * lda + gk + q];
            As[buf][r][c+q] = vq;
          }
        }
      }
#endif
    } else {
      for (int e = tid * 4; e < BM * BK; e += NT * 4) {
        int r = e >> 5, c = e & 31;
        int gm = bm0 + r, gk = k0 + c;
        float4 v = make_float4(0.f, 0.f, 0.f, 0.f);
        if (gm < M) {
          if (gk + 3 < K) {
            v = *(const float4*)(Af + (long)gm * lda + gk);
          } else {
            float* pv = &v.x;
#pragma unroll
            for (int q = 0; q < 4; ++q)
              if (gk + q < K) pv[q] = Af[(long)gm * lda + gk + q];
          }
        }
        As[buf][r][c]   = (__bf16)v.x; As[buf][r][c+1] = (__bf16)v.y;
        As[buf][r][c+2] = (__bf16)v.z; As[buf][r][c+3] = (__bf16)v.w;
      }
    }
  };

  v8f acc[MFRAG][NFRAG];
  v8f vzero = {};
#pragma unroll
  for (int i = 0; i < MFRAG; ++i)
#pragma unroll
    for (int j = 0; j < NFRAG; ++j) acc[i][j] = vzero;

  const int nT = (K + BK - 1) / BK;
  stage_a(0, 0);   // pipeline preamble

  for (int it = 0; it < nT; ++it) {
    const int k0  = it * BK;
    const int buf = it & 1;

    // ---- stage B tile (32 x BN), f32 -> bf16, transposed into LDS ----
    for (int e = tid * 4; e < BK * BN; e += NT * 4) {
      int kk = e / BN, nn = e % BN;
      int gk = k0 + kk, gn = bn0 + nn;
      float4 v = make_float4(0.f, 0.f, 0.f, 0.f);
      if (gk < K) {
        if (gn + 3 < N) {
          v = *(const float4*)(Bz + (long)gk * ldb + gn);
          if (gk + BK < K)   // prefetch next K tile of B (global_prefetch_b8)
            __builtin_prefetch((const void*)(Bz + (long)(gk + BK) * ldb + gn), 0, 1);
        } else {
          float* pv = &v.x;
#pragma unroll
          for (int q = 0; q < 4; ++q)
            if (gn + q < N) pv[q] = Bz[(long)gk * ldb + gn + q];
        }
      }
      Bs[nn  ][kk] = (__bf16)v.x; Bs[nn+1][kk] = (__bf16)v.y;
      Bs[nn+2][kk] = (__bf16)v.z; Bs[nn+3][kk] = (__bf16)v.w;
    }

    // ---- pipeline: launch A tile it+1 into the other buffer, then make
    //      sure tile it has landed (<=1 outstanding TDM op) ----
    if (it + 1 < nT) {
      stage_a(it + 1, buf ^ 1);
#if HAVE_TDM
      if (ABF16 && wave == 0) __builtin_amdgcn_s_wait_tensorcnt(1);
#endif
    } else {
#if HAVE_TDM
      if (ABF16 && wave == 0) __builtin_amdgcn_s_wait_tensorcnt(0);
#endif
    }
    __syncthreads();

    // ---- fragments per ISA 7.12.2 ----
    // A 16x32: lane l -> row m=l&15, kbase=(l>>4)*8; k = kbase+{0..7}, kbase+16+{0..7}
    v16bf af[MFRAG];
    {
      int am  = lane & 15;
      int akb = (lane >> 4) * 8;
#pragma unroll
      for (int i = 0; i < MFRAG; ++i) {
        const __bf16* p = &As[buf][wm*MFRAG*16 + i*16 + am][akb];
        v8bf lo = *(const v8bf*)p;
        v8bf hi = *(const v8bf*)(p + 16);
#pragma unroll
        for (int q = 0; q < 8; ++q) { af[i][q] = lo[q]; af[i][q+8] = hi[q]; }
      }
    }
    // B 32x16: lane l -> col n=l&15, kbase=(l>>4)*8; k = kbase+{0..7}, kbase+16+{0..7}
    v16bf bf_[NFRAG];
    {
      int bn_ = lane & 15;
      int bkb = (lane >> 4) * 8;
#pragma unroll
      for (int j = 0; j < NFRAG; ++j) {
        const __bf16* p = &Bs[wn*NFRAG*16 + j*16 + bn_][bkb];
        v8bf lo = *(const v8bf*)p;
        v8bf hi = *(const v8bf*)(p + 16);
#pragma unroll
        for (int q = 0; q < 8; ++q) { bf_[j][q] = lo[q]; bf_[j][q+8] = hi[q]; }
      }
    }
#pragma unroll
    for (int i = 0; i < MFRAG; ++i)
#pragma unroll
      for (int j = 0; j < NFRAG; ++j)
        acc[i][j] = __builtin_amdgcn_wmma_f32_16x16x32_bf16(
            false, af[i], false, bf_[j], (short)0, acc[i][j], false, false);
    __syncthreads();   // protects Bs and the buffer TDM targets next iter
  }

  // ---- epilogue: C/D layout: lane n=l&15; vgpr r -> m = r + 8*(l>>4) ----
  {
    int cn  = lane & 15;
    int cm8 = (lane >> 4) * 8;
#pragma unroll
    for (int i = 0; i < MFRAG; ++i)
#pragma unroll
      for (int j = 0; j < NFRAG; ++j) {
        int col = bn0 + wn*NFRAG*16 + j*16 + cn;
        if (col >= N) continue;
        float bv = bias ? bias[col] : 0.f;
#pragma unroll
        for (int r = 0; r < 8; ++r) {
          int m = bm0 + wm*MFRAG*16 + i*16 + r + cm8;
          if (m < M) {
            float val = acc[i][j][r] + bv;
            if (ACT == 1) val = val > 0.f ? val : 0.f;
            if (CBF16) Czh[(long)m * ldc + col] = (__bf16)val;
            else       Czf[(long)m * ldc + col] = val;
          }
        }
      }
  }
}

// =====================================================================
// Embedding gather + in_proj (K=70 GEMV per token). block = 320 threads.
// Writes f32 master + bf16 mirror.
// =====================================================================
__global__ void embed_inproj_kernel(const int* __restrict__ atype,
                                    const float* __restrict__ emb,
                                    const float* __restrict__ xc,
                                    const float* __restrict__ xv,
                                    const float* __restrict__ W,
                                    const float* __restrict__ bias,
                                    float* __restrict__ feats,
                                    unsigned short* __restrict__ featsb)
{
  int t = blockIdx.x;
  __shared__ float xin[70];
  int c = threadIdx.x;
  if (c < 64)      xin[c] = emb[atype[t] * 64 + c];
  else if (c < 67) xin[c] = xc[t * 3 + (c - 64)];
  else if (c < 70) xin[c] = xv[t * 3 + (c - 67)];
  __syncthreads();
  float acc = bias[c];
#pragma unroll 5
  for (int j = 0; j < 70; ++j) acc += xin[j] * W[j * HID + c];
  feats[(long)t * HID + c] = acc;
  ((__bf16*)featsb)[(long)t * HID + c] = (__bf16)acc;
}

// =====================================================================
// Gaussian attention weights, normalized, stored bf16: wts[(h*4+b)][i][j]
// =====================================================================
__global__ void wts_kernel(const float* __restrict__ xc, unsigned short* __restrict__ wraw)
{
  __bf16* wts = reinterpret_cast<__bf16*>(wraw);
  int bi = blockIdx.x;
  int b = bi >> 11, i = bi & (NTOK - 1);
  __shared__ float d2row[NTOK];
  __shared__ float red[5][8];
  __shared__ float inv_sum[5];
  const float inv_ls2[5] = {100.f, 25.f, 4.f, 1.f, 0.25f};

  const float* base = xc + (long)b * NTOK * 3;
  float xi0 = base[i*3], xi1 = base[i*3+1], xi2 = base[i*3+2];
  float sums[5] = {0.f, 0.f, 0.f, 0.f, 0.f};
  for (int j = threadIdx.x; j < NTOK; j += 256) {
    float d0 = xi0 - base[j*3], d1 = xi1 - base[j*3+1], d2 = xi2 - base[j*3+2];
    float dd = d0*d0 + d1*d1 + d2*d2;
    d2row[j] = dd;
#pragma unroll
    for (int h = 0; h < 5; ++h) sums[h] += __expf(-dd * inv_ls2[h]);
  }
  int lane = threadIdx.x & 31, w = threadIdx.x >> 5;
#pragma unroll
  for (int h = 0; h < 5; ++h) {
    float s = sums[h];
    for (int o = 16; o > 0; o >>= 1) s += __shfl_down(s, o, 32);
    if (lane == 0) red[h][w] = s;
  }
  __syncthreads();
  if (threadIdx.x < 5) {
    float tot = 0.f;
#pragma unroll
    for (int q = 0; q < 8; ++q) tot += red[threadIdx.x][q];
    inv_sum[threadIdx.x] = 1.f / (tot + 1e-8f);
  }
  __syncthreads();
#pragma unroll
  for (int h = 0; h < 5; ++h) {
    __bf16* wrow = wts + (((long)(h * NBATCH + b) * NTOK + i) * NTOK);
    float il = inv_ls2[h], is = inv_sum[h];
    for (int j = threadIdx.x; j < NTOK; j += 256)
      wrow[j] = (__bf16)(__expf(-d2row[j] * il) * is);
  }
}

// repack vproj_w (5,320,64) -> (320, 320) row-major, cols = h*64+d
__global__ void repack_vw_kernel(const float* __restrict__ vw, float* __restrict__ out)
{
  int idx = blockIdx.x * blockDim.x + threadIdx.x;
  if (idx >= HID * HID) return;
  int f = idx / HID, n = idx % HID;
  int h = n >> 6, d = n & 63;
  out[idx] = vw[((long)h * HID + f) * 64 + d];
}

// feats = LayerNorm(feats + delta)*g + b; also refresh bf16 mirror.
__global__ void resid_ln_kernel(float* __restrict__ feats, unsigned short* __restrict__ featsb,
                                const float* __restrict__ delta,
                                const float* __restrict__ g, const float* __restrict__ b)
{
  int t = blockIdx.x, c = threadIdx.x;
  __shared__ float red[10];
  __shared__ float mu_s, var_s;
  float v = feats[(long)t * HID + c] + delta[(long)t * HID + c];
  float s = v;
  for (int o = 16; o > 0; o >>= 1) s += __shfl_down(s, o, 32);
  if ((c & 31) == 0) red[c >> 5] = s;
  __syncthreads();
  if (c == 0) {
    float tot = 0.f;
#pragma unroll
    for (int q = 0; q < 10; ++q) tot += red[q];
    mu_s = tot * (1.f / HID);
  }
  __syncthreads();
  float d = v - mu_s;
  float s2 = d * d;
  for (int o = 16; o > 0; o >>= 1) s2 += __shfl_down(s2, o, 32);
  if ((c & 31) == 0) red[c >> 5] = s2;
  __syncthreads();
  if (c == 0) {
    float tot = 0.f;
#pragma unroll
    for (int q = 0; q < 10; ++q) tot += red[q];
    var_s = tot * (1.f / HID);
  }
  __syncthreads();
  float out = d * rsqrtf(var_s + 1e-5f) * g[c] + b[c];
  feats[(long)t * HID + c] = out;
  ((__bf16*)featsb)[(long)t * HID + c] = (__bf16)out;
}

// CI (bf16, width 352): [0:3]=y_coords, [3:323]=feats, [323:352]=0
__global__ void build_ci_kernel(const float* __restrict__ ycoords,
                                const float* __restrict__ feats,
                                unsigned short* __restrict__ CIb)
{
  int idx = blockIdx.x * blockDim.x + threadIdx.x;
  if (idx >= TOK * CIW) return;
  int t = idx / CIW, c = idx % CIW;
  float v = 0.f;
  if (c < 3)        v = ycoords[t * 3 + c];
  else if (c < 323) v = feats[(long)t * HID + (c - 3)];
  ((__bf16*)CIb)[idx] = (__bf16)v;
}

// out[t][c] = bias[c] + sum_k H[t][k] * W[k][c], c<3, H is bf16
__global__ void proj3_kernel(const unsigned short* __restrict__ H, const float* __restrict__ W,
                             const float* __restrict__ bias, float* __restrict__ out, int T)
{
  int idx = blockIdx.x * blockDim.x + threadIdx.x;
  if (idx >= T * 3) return;
  int t = idx / 3, c = idx % 3;
  float acc = bias[c];
  const __bf16* h = (const __bf16*)H + (long)t * HID;
#pragma unroll 4
  for (int k = 0; k < HID; ++k) acc += (float)h[k] * W[k * 3 + c];
  out[idx] = acc;
}

__global__ void init_flow_kernel(const float* __restrict__ yv, float* __restrict__ vel,
                                 float* __restrict__ ld)
{
  int t = blockIdx.x * blockDim.x + threadIdx.x;
  if (t >= TOK) return;
  ld[t] = 0.f;
  vel[t*3] = yv[t*3]; vel[t*3+1] = yv[t*3+1]; vel[t*3+2] = yv[t*3+2];
}

__global__ void flow_update_kernel(float* __restrict__ vel, const float* __restrict__ S,
                                   const float* __restrict__ Tb, float* __restrict__ ld)
{
  int t = blockIdx.x * blockDim.x + threadIdx.x;
  if (t >= TOK) return;
  float acc = 0.f;
#pragma unroll
  for (int c = 0; c < 3; ++c) {
    float s = S[t*3 + c];
    vel[t*3 + c] = vel[t*3 + c] * __expf(s) + Tb[t*3 + c];
    acc += s;
  }
  ld[t] += acc;
}

// out = [z_coords(24576) | z_velocs(24576) | logp(8192)]
__global__ void final_kernel(const float* __restrict__ yc, const float* __restrict__ vel,
                             const float* __restrict__ ld, const float* __restrict__ lsc,
                             const float* __restrict__ lsv, float* __restrict__ out)
{
  int t = blockIdx.x * blockDim.x + threadIdx.x;
  if (t >= TOK) return;
  float isc = __expf(-lsc[0]), isv = __expf(-lsv[0]);
  float qa = 0.f, qb = 0.f;
#pragma unroll
  for (int c = 0; c < 3; ++c) {
    float zc = yc[t*3 + c];
    float zv = vel[t*3 + c];
    out[t*3 + c] = zc;
    out[TOK*3 + t*3 + c] = zv;
    float a = zc * isc, b = zv * isv;
    qa += a * a; qb += b * b;
  }
  out[TOK*6 + t] = -0.5f * qa - 0.5f * qb + ld[t];
}

// =====================================================================
// Host orchestration. Inputs: setup_inputs() dict insertion order, depth-first:
//   0 embedding, 1 in_proj_w, 2 in_proj_b,
//   3+blk*12: vproj_w, vproj_b, out_w, out_b, ln1_g, ln1_b, ln2_g, ln2_b,
//             ffn_w1, ffn_b1, ffn_w2, ffn_b2          (blk=0..3)
//   51+fl*12: scale{w0,b0,w1,b1,w2,b2}, shift{...}    (fl=0..3)
//   99 log_scale_coords, 100 log_scale_velocs,
//   101 atom_types, 102 x_coords, 103 x_velocs, 104 y_coords, 105 y_velocs
// =====================================================================
extern "C" void kernel_launch(void* const* d_in, const int* in_sizes, int n_in,
                              void* d_out, int out_size, void* d_ws, size_t ws_size,
                              hipStream_t stream)
{
  (void)in_sizes; (void)n_in; (void)out_size; (void)ws_size;
  const float* emb   = (const float*)d_in[0];
  const float* inW   = (const float*)d_in[1];
  const float* inB   = (const float*)d_in[2];
  const float* lsc   = (const float*)d_in[99];
  const float* lsv   = (const float*)d_in[100];
  const int*   atype = (const int*)  d_in[101];
  const float* xc    = (const float*)d_in[102];
  const float* xv    = (const float*)d_in[103];
  const float* yc    = (const float*)d_in[104];
  const float* yv    = (const float*)d_in[105];

  size_t off = 0;
  auto carve = [&](size_t bytes) -> void* {
    void* p = (char*)d_ws + off;
    off += (bytes + 255) & ~(size_t)255;
    return p;
  };
  unsigned short* wts    = (unsigned short*)carve((size_t)5 * NBATCH * NTOK * NTOK * 2);
  float*          feats  = (float*)         carve((size_t)TOK * HID * 4);
  unsigned short* featsb = (unsigned short*)carve((size_t)TOK * HID * 2);
  float*          Vbuf   = (float*)         carve((size_t)TOK * HID * 4);
  unsigned short* attb   = (unsigned short*)carve((size_t)TOK * HID * 2);
  float*          t320   = (float*)         carve((size_t)TOK * HID * 4);
  unsigned short* tbig   = (unsigned short*)carve((size_t)TOK * 1280 * 2);
  float*          WvP    = (float*)         carve((size_t)HID * HID * 4);
  unsigned short* CIb    = (unsigned short*)carve((size_t)TOK * CIW * 2);
  unsigned short* h1     = (unsigned short*)carve((size_t)TOK * HID * 2);
  unsigned short* h2     = (unsigned short*)carve((size_t)TOK * HID * 2);
  float*          sbuf   = (float*)         carve((size_t)TOK * 3 * 4);
  float*          tshift = (float*)         carve((size_t)TOK * 3 * 4);
  float*          vel    = (float*)         carve((size_t)TOK * 3 * 4);
  float*          ldet   = (float*)         carve((size_t)TOK * 4);

  // ---- input features + (block-invariant) attention weights ----
  embed_inproj_kernel<<<TOK, HID, 0, stream>>>(atype, emb, xc, xv, inW, inB, feats, featsb);
  wts_kernel<<<NBATCH * NTOK, 256, 0, stream>>>(xc, wts);

  // ---- transformer blocks ----
  for (int blk = 0; blk < 4; ++blk) {
    int p = 3 + blk * 12;
    const float* vw  = (const float*)d_in[p+0];
    const float* vb  = (const float*)d_in[p+1];
    const float* ow  = (const float*)d_in[p+2];
    const float* ob  = (const float*)d_in[p+3];
    const float* l1g = (const float*)d_in[p+4];
    const float* l1b = (const float*)d_in[p+5];
    const float* l2g = (const float*)d_in[p+6];
    const float* l2b = (const float*)d_in[p+7];
    const float* fw1 = (const float*)d_in[p+8];
    const float* fb1 = (const float*)d_in[p+9];
    const float* fw2 = (const float*)d_in[p+10];
    const float* fb2 = (const float*)d_in[p+11];

    repack_vw_kernel<<<(HID*HID + 255)/256, 256, 0, stream>>>(vw, WvP);
    // V = feats @ WvP + vb   (8192x320x320) -> f32 (B operand of attention)
    gemm_wmma<4,2,2,4,true,0,0><<<dim3(3, 64, 1), 256, 0, stream>>>(
        (const void*)featsb, WvP, vb, (void*)Vbuf, TOK, HID, HID, HID, HID, HID, 0, 0, 0);
    // att[h,b] = wts[h,b] @ V[h,b]  (2048x64x2048, A bf16 via pipelined TDM, C bf16)
    for (int h = 0; h < 5; ++h) {
      gemm_wmma<4,2,2,2,true,0,1><<<dim3(1, NTOK/128, NBATCH), 256, 0, stream>>>(
          (const void*)(wts + (size_t)h * NBATCH * NTOK * NTOK), Vbuf + h*64, nullptr,
          (void*)(attb + h*64), NTOK, 64, NTOK, NTOK, HID, HID,
          (long)NTOK * NTOK, (long)NTOK * HID, (long)NTOK * HID);
    }
    // out proj (C f32 for residual) + LN1
    gemm_wmma<4,2,2,4,true,0,0><<<dim3(3, 64, 1), 256, 0, stream>>>(
        (const void*)attb, ow, ob, (void*)t320, TOK, HID, HID, HID, HID, HID, 0, 0, 0);
    resid_ln_kernel<<<TOK, HID, 0, stream>>>(feats, featsb, t320, l1g, l1b);
    // FFN: hidden stored bf16 (consumed as bf16 anyway), out f32 + LN2
    gemm_wmma<4,2,2,4,true,1,1><<<dim3(10, 64, 1), 256, 0, stream>>>(
        (const void*)featsb, fw1, fb1, (void*)tbig, TOK, 1280, HID, HID, 1280, 1280, 0, 0, 0);
    gemm_wmma<4,2,2,4,true,0,0><<<dim3(3, 64, 1), 256, 0, stream>>>(
        (const void*)tbig, fw2, fb2, (void*)t320, TOK, HID, 1280, 1280, HID, HID, 0, 0, 0);
    resid_ln_kernel<<<TOK, HID, 0, stream>>>(feats, featsb, t320, l2g, l2b);
  }

  // ---- flows: CI fixed (y[:, :3] never changes), only velocities update ----
  build_ci_kernel<<<(TOK*CIW + 255)/256, 256, 0, stream>>>(yc, feats, CIb);
  init_flow_kernel<<<(TOK + 255)/256, 256, 0, stream>>>(yv, vel, ldet);
  for (int fl = 0; fl < 4; ++fl) {
    int p = 51 + fl * 12;
    for (int half = 0; half < 2; ++half) {
      const float* w0 = (const float*)d_in[p + half*6 + 0];
      const float* b0 = (const float*)d_in[p + half*6 + 1];
      const float* w1 = (const float*)d_in[p + half*6 + 2];
      const float* b1 = (const float*)d_in[p + half*6 + 3];
      const float* w2 = (const float*)d_in[p + half*6 + 4];
      const float* b2 = (const float*)d_in[p + half*6 + 5];
      float* out3 = half ? tshift : sbuf;
      gemm_wmma<4,2,2,4,true,1,1><<<dim3(3, 64, 1), 256, 0, stream>>>(
          (const void*)CIb, w0, b0, (void*)h1, TOK, HID, 323, CIW, HID, HID, 0, 0, 0);
      gemm_wmma<4,2,2,4,true,1,1><<<dim3(3, 64, 1), 256, 0, stream>>>(
          (const void*)h1, w1, b1, (void*)h2, TOK, HID, HID, HID, HID, HID, 0, 0, 0);
      proj3_kernel<<<(TOK*3 + 255)/256, 256, 0, stream>>>(h2, w2, b2, out3, TOK);
    }
    flow_update_kernel<<<(TOK + 255)/256, 256, 0, stream>>>(vel, sbuf, tshift, ldet);
  }
  final_kernel<<<(TOK + 255)/256, 256, 0, stream>>>(yc, vel, ldet, lsc, lsv, (float*)d_out);
}
